// Expert_58583353917770
// MI455X (gfx1250) — matrix operand order repaired and checked by reading, not compiled, not run on hardware
//
#include <hip/hip_runtime.h>
#include <hip/hip_bf16.h>

// ---------------- model dims ----------------
#define DMODEL   2048
#define NHEADS   8
#define HDIM     256
#define DINNER   4096
#define DSTATE   16
#define DCONV    4
#define DTRANK   128
#define BATCH    2
#define SEQ      2048
#define BL       (BATCH * SEQ)      // 4096 rows for [B,L,*] matrices

// ---------------- types ----------------
typedef __attribute__((ext_vector_type(8)))  float         v8f;
typedef __attribute__((ext_vector_type(16))) __bf16        v16bf;
typedef __attribute__((ext_vector_type(4)))  __bf16        bf16x4;
typedef __attribute__((ext_vector_type(4)))  unsigned int  u32x4;
typedef __attribute__((ext_vector_type(2)))  unsigned int  u32x2;
typedef __attribute__((ext_vector_type(4)))  float         f32x4;

union FragU  { u32x4 u[2]; v16bf v; };
union Pack4  { bf16x4 h; u32x2 u; };

// ---------------- generic WMMA GEMM ----------------
// C[m,n] = epilogue( sum_k A[m,k] * W[n,k] )
// A element at A[m*lda + k]; W element at W[n*ldwn + k*ldwk]
// REQUIRES: M % 128 == 0, K % 32 == 0 (true for every launch here).
// batching over gridDim.z: z -> (zb = z/nh, zh = z%nh), per-matrix offsets.
struct GemmParams {
  const float* A;  long long lda;
  const float* W;  long long ldwn, ldwk;
  float*       C;  long long ldc;
  const float* bias;
  const float* resid; long long ldr;
  int M, N, K;
  float scale;
  int flags;          // 1=bias, 2=scale, 4=residual add, 8=softplus
  int nh;
  long long offAb, offAh, offWb, offWh, offCb, offCh;
};

#define BM 128
#define BN 128
#define BK 32
#define LDSS 40   // ushorts per LDS row (BK + 8 pad -> 80B row stride, 16B aligned)

struct StageRegs { f32x4 a[4]; f32x4 w[4]; };

// phase 1: issue all global loads for the next tile (no LDS stores yet)
template<bool VECW>
__device__ __forceinline__ void load_regs(const GemmParams& p,
                                          const float* __restrict__ A,
                                          const float* __restrict__ W,
                                          int bm0, int bn0, int k0, int tid,
                                          StageRegs& sr) {
  #pragma unroll
  for (int j = 0; j < 4; ++j) {
    int lin = tid + j * 256;              // 0..1023
    int r   = lin >> 3;                   // 0..127
    int c   = (lin & 7) << 2;             // 0,4,...,28
    sr.a[j] = *(const f32x4*)(A + (long long)(bm0 + r) * p.lda + (k0 + c));
  }
  #pragma unroll
  for (int j = 0; j < 4; ++j) {
    int lin = tid + j * 256;
    int r   = lin >> 3;
    int c   = (lin & 7) << 2;
    int gn  = bn0 + r;
    f32x4 v = {0.f, 0.f, 0.f, 0.f};
    if (gn < p.N) {                       // N guard only (x_proj has N = 160)
      const float* src = W + (long long)gn * p.ldwn + (long long)(k0 + c) * p.ldwk;
      if (VECW) v = *(const f32x4*)src;
      else { v.x = src[0]; v.y = src[p.ldwk]; v.z = src[2 * p.ldwk]; v.w = src[3 * p.ldwk]; }
    }
    sr.w[j] = v;
  }
}

// phase 2: packed f32->bf16 convert + LDS store (placed after the WMMAs)
__device__ __forceinline__ void store_tiles(int tid, const StageRegs& sr,
                                            unsigned short* Adst, unsigned short* Wdst) {
  #pragma unroll
  for (int j = 0; j < 4; ++j) {
    int lin = tid + j * 256;
    int r   = lin >> 3;
    int c   = (lin & 7) << 2;
    Pack4 pk; pk.h = __builtin_convertvector(sr.a[j], bf16x4);
    *(u32x2*)&Adst[r * LDSS + c] = pk.u;
  }
  #pragma unroll
  for (int j = 0; j < 4; ++j) {
    int lin = tid + j * 256;
    int r   = lin >> 3;
    int c   = (lin & 7) << 2;
    Pack4 pk; pk.h = __builtin_convertvector(sr.w[j], bf16x4);
    *(u32x2*)&Wdst[r * LDSS + c] = pk.u;
  }
}

template<bool VECW>
__global__ __launch_bounds__(256) void wmma_gemm(GemmParams p) {
  __shared__ unsigned short As[2][BM * LDSS];
  __shared__ unsigned short Ws[2][BN * LDSS];

  const int tid  = threadIdx.x;
  const int lane = tid & 31;
  const int wave = tid >> 5;       // 0..7
  const int wm   = wave >> 2;      // 0..1  (64 rows each)
  const int wn   = wave & 3;       // 0..3  (32 cols each)

  const int z  = blockIdx.z;
  const int zb = z / p.nh, zh = z % p.nh;
  const float* A = p.A + (long long)zb * p.offAb + (long long)zh * p.offAh;
  const float* W = p.W + (long long)zb * p.offWb + (long long)zh * p.offWh;
  float*       C = p.C + (long long)zb * p.offCb + (long long)zh * p.offCh;

  const int bm0 = blockIdx.y * BM;
  const int bn0 = blockIdx.x * BN;

  v8f acc[4][2];
  const v8f vzero = {0.f,0.f,0.f,0.f,0.f,0.f,0.f,0.f};
  #pragma unroll
  for (int i = 0; i < 4; ++i)
    #pragma unroll
    for (int j = 0; j < 2; ++j)
      acc[i][j] = vzero;

  const int mrow16 = lane & 15;
  const int khalf  = lane >> 4;
  const int nk = p.K / BK;

  // prologue: stage tile 0 into buffer 0
  StageRegs sr;
  load_regs<VECW>(p, A, W, bm0, bn0, 0, tid, sr);
  store_tiles(tid, sr, As[0], Ws[0]);

  for (int t = 0; t < nk; ++t) {
    __syncthreads();   // stage t's LDS writes visible; prior reads of this buffer done

    const bool more = (t + 1 < nk);
    // issue next tile's global loads immediately (latency hidden by WMMAs below)
    if (more)
      load_regs<VECW>(p, A, W, bm0, bn0, (t + 1) * BK, tid, sr);
    // keep L2 warm two tiles ahead (gfx1250 global_prefetch)
    if (t + 2 < nk) {
      int r = tid >> 3, c = (tid & 7) << 2;
      __builtin_prefetch(A + (long long)(bm0 + r) * p.lda + ((t + 2) * BK + c), 0, 1);
    }

    const unsigned short* Ab = As[t & 1];
    const unsigned short* Wb = Ws[t & 1];

    // A 16x32 bf16 frag: lane l (m=l&15): elems 0..7 = A[m][kh*8+0..7], 8..15 = A[m][kh*8+16..23]
    FragU fa[4];
    #pragma unroll
    for (int ti = 0; ti < 4; ++ti) {
      const unsigned short* base = &Ab[(wm * 64 + ti * 16 + mrow16) * LDSS];
      fa[ti].u[0] = *(const u32x4*)(base + khalf * 8);
      fa[ti].u[1] = *(const u32x4*)(base + khalf * 8 + 16);
    }
    // B 32x16 bf16 frag: lane l (n=l&15): elems 0..7 = B[kh*8+0..7][n], 8..15 = B[kh*8+16..23][n]  (Ws is [n][k])
    FragU fb[2];
    #pragma unroll
    for (int tj = 0; tj < 2; ++tj) {
      const unsigned short* base = &Wb[(wn * 32 + tj * 16 + mrow16) * LDSS];
      fb[tj].u[0] = *(const u32x4*)(base + khalf * 8);
      fb[tj].u[1] = *(const u32x4*)(base + khalf * 8 + 16);
    }

    #pragma unroll
    for (int ti = 0; ti < 4; ++ti)
      #pragma unroll
      for (int tj = 0; tj < 2; ++tj)
        acc[ti][tj] = __builtin_amdgcn_wmma_f32_16x16x32_bf16(
            false, fa[ti].v, false, fb[tj].v, (short)0, acc[ti][tj], false, false);

    // after the math: wait for next tile's loads, convert, store to other buffer
    if (more)
      store_tiles(tid, sr, As[(t + 1) & 1], Ws[(t + 1) & 1]);
  }

  // ---- epilogue: D layout: VGPR j -> row j + 8*(lane>>4), col = lane&15 ----
  #pragma unroll
  for (int ti = 0; ti < 4; ++ti) {
    #pragma unroll
    for (int tj = 0; tj < 2; ++tj) {
      int n = bn0 + wn * 32 + tj * 16 + (lane & 15);
      if (n >= p.N) continue;
      float bv = (p.flags & 1) ? p.bias[n] : 0.f;
      #pragma unroll
      for (int j = 0; j < 8; ++j) {
        int m = bm0 + wm * 64 + ti * 16 + j + 8 * khalf;
        float v = acc[ti][tj][j];
        if (p.flags & 2) v *= p.scale;
        v += bv;
        if (p.flags & 4) v += p.resid[(long long)m * p.ldr + n];
        if (p.flags & 8) v = (v > 20.f) ? v : log1pf(__expf(v));
        C[(long long)m * p.ldc + n] = v;
      }
    }
  }
}

// ---------------- row softmax (in-place) ----------------
__global__ __launch_bounds__(256) void softmax_rows(float* S, int ncols) {
  __shared__ float red[256];
  const int tid = threadIdx.x;
  float* p = S + (long long)blockIdx.x * ncols;

  float m = -3.0e38f;
  for (int i = tid; i < ncols; i += 256) m = fmaxf(m, p[i]);
  red[tid] = m; __syncthreads();
  for (int s = 128; s > 0; s >>= 1) { if (tid < s) red[tid] = fmaxf(red[tid], red[tid + s]); __syncthreads(); }
  m = red[0]; __syncthreads();

  float sum = 0.f;
  for (int i = tid; i < ncols; i += 256) { float e = __expf(p[i] - m); p[i] = e; sum += e; }
  red[tid] = sum; __syncthreads();
  for (int s = 128; s > 0; s >>= 1) { if (tid < s) red[tid] += red[tid + s]; __syncthreads(); }
  float inv = 1.f / red[0];
  for (int i = tid; i < ncols; i += 256) p[i] *= inv;
}

// ---------------- LayerNorm ----------------
__global__ __launch_bounds__(256) void layernorm_rows(const float* X, float* Y,
                                                      const float* g, const float* b, int ncols) {
  __shared__ float red[256];
  const int tid = threadIdx.x;
  const float* x = X + (long long)blockIdx.x * ncols;
  float*       y = Y + (long long)blockIdx.x * ncols;

  float s = 0.f;
  for (int i = tid; i < ncols; i += 256) s += x[i];
  red[tid] = s; __syncthreads();
  for (int k = 128; k > 0; k >>= 1) { if (tid < k) red[tid] += red[tid + k]; __syncthreads(); }
  float mean = red[0] / ncols; __syncthreads();

  float v = 0.f;
  for (int i = tid; i < ncols; i += 256) { float d = x[i] - mean; v += d * d; }
  red[tid] = v; __syncthreads();
  for (int k = 128; k > 0; k >>= 1) { if (tid < k) red[tid] += red[tid + k]; __syncthreads(); }
  float inv = rsqrtf(red[0] / ncols + 1e-5f);
  for (int i = tid; i < ncols; i += 256)
    y[i] = (x[i] - mean) * inv * g[i] + b[i];
}

// ---------------- depthwise causal conv1d + SiLU ----------------
__global__ __launch_bounds__(256) void conv_silu(const float* xz, const float* cw,
                                                 const float* cb, float* uc) {
  long long idx = (long long)blockIdx.x * 256 + threadIdx.x;  // over BL*DINNER
  int d = (int)(idx % DINNER);
  long long bl = idx / DINNER;
  int l = (int)(bl % SEQ);
  long long b = bl / SEQ;

  float acc = cb[d];
  #pragma unroll
  for (int j = 0; j < DCONV; ++j) {
    int ll = l - (DCONV - 1) + j;
    if (ll >= 0)
      acc += xz[((b * SEQ + ll) * (long long)(2 * DINNER)) + d] * cw[d * DCONV + j];
  }
  uc[idx] = acc / (1.f + __expf(-acc));   // silu
}

// ---------------- selective scan (fuses +u*D and *silu(z)) ----------------
__global__ __launch_bounds__(256) void sel_scan(const float* uc, const float* delta,
                                                const float* xdbl, const float* A_log,
                                                const float* Dp, const float* xz, float* y) {
  const int b  = blockIdx.x / (DINNER / 256);
  const int d  = (blockIdx.x % (DINNER / 256)) * 256 + threadIdx.x;
  const int tid = threadIdx.x;

  float A[DSTATE], h[DSTATE];
  #pragma unroll
  for (int s = 0; s < DSTATE; ++s) {
    A[s] = -__expf(A_log[(long long)d * DSTATE + s]);
    h[s] = 0.f;
  }
  const float Dd = Dp[d];

  __shared__ float Bs[DSTATE];
  __shared__ float Cs[DSTATE];

  for (int l = 0; l < SEQ; ++l) {
    long long row = (long long)b * SEQ + l;
    if (tid < DSTATE)            Bs[tid]          = xdbl[row * (DTRANK + 2 * DSTATE) + DTRANK + tid];
    else if (tid < 2 * DSTATE)   Cs[tid - DSTATE] = xdbl[row * (DTRANK + 2 * DSTATE) + DTRANK + tid];
    __syncthreads();

    float dt = delta[row * DINNER + d];
    float u  = uc[row * DINNER + d];
    float du = dt * u;
    float yv = 0.f;
    #pragma unroll
    for (int s = 0; s < DSTATE; ++s) {
      h[s] = h[s] * __expf(dt * A[s]) + du * Bs[s];
      yv += h[s] * Cs[s];
    }
    float zv = xz[row * (long long)(2 * DINNER) + DINNER + d];
    y[row * DINNER + d] = (yv + u * Dd) * (zv / (1.f + __expf(-zv)));
    __syncthreads();
  }
}

// ---------------- host side ----------------
static void gemm(hipStream_t st,
                 const float* A, long long lda,
                 const float* W, long long ldwn, long long ldwk,
                 float* C, long long ldc,
                 const float* bias, const float* resid, long long ldr,
                 int M, int N, int K, float scale, int flags,
                 int Z = 1, int nh = 1,
                 long long oAb = 0, long long oAh = 0,
                 long long oWb = 0, long long oWh = 0,
                 long long oCb = 0, long long oCh = 0) {
  GemmParams p;
  p.A = A; p.lda = lda; p.W = W; p.ldwn = ldwn; p.ldwk = ldwk;
  p.C = C; p.ldc = ldc; p.bias = bias; p.resid = resid; p.ldr = ldr;
  p.M = M; p.N = N; p.K = K; p.scale = scale; p.flags = flags;
  p.nh = nh; p.offAb = oAb; p.offAh = oAh; p.offWb = oWb; p.offWh = oWh;
  p.offCb = oCb; p.offCh = oCh;
  dim3 grid((N + BN - 1) / BN, (M + BM - 1) / BM, Z);
  if (ldwk == 1) wmma_gemm<true ><<<grid, 256, 0, st>>>(p);
  else           wmma_gemm<false><<<grid, 256, 0, st>>>(p);
}

extern "C" void kernel_launch(void* const* d_in, const int* in_sizes, int n_in,
                              void* d_out, int out_size, void* d_ws, size_t ws_size,
                              hipStream_t stream) {
  const float* x          = (const float*)d_in[0];
  const float* proj_w     = (const float*)d_in[1];
  const float* proj_b     = (const float*)d_in[2];
  const float* qkv_w      = (const float*)d_in[3];
  const float* qkv_b      = (const float*)d_in[4];
  const float* attn_out_w = (const float*)d_in[5];
  const float* attn_out_b = (const float*)d_in[6];
  const float* ln_g       = (const float*)d_in[7];
  const float* ln_b       = (const float*)d_in[8];
  const float* in_proj_w  = (const float*)d_in[9];
  const float* conv_w     = (const float*)d_in[10];
  const float* conv_b     = (const float*)d_in[11];
  const float* x_proj_w   = (const float*)d_in[12];
  const float* dt_proj_w  = (const float*)d_in[13];
  const float* dt_proj_b  = (const float*)d_in[14];
  const float* A_log      = (const float*)d_in[15];
  const float* Dvec       = (const float*)d_in[16];
  const float* out_proj_w = (const float*)d_in[17];
  float* out = (float*)d_out;

  // workspace layout (floats), total ~810 MB
  float* ws        = (float*)d_ws;
  float* buf_h     = ws;                                              // BL*DMODEL
  float* buf_qkv   = buf_h     + (size_t)BL * DMODEL;                 // BL*3*DMODEL
  float* buf_sc    = buf_qkv   + (size_t)BL * 3 * DMODEL;             // B*H*SEQ*SEQ
  float* buf_ctx   = buf_sc    + (size_t)BATCH * NHEADS * SEQ * SEQ;  // BL*DMODEL
  float* buf_attn  = buf_ctx   + (size_t)BL * DMODEL;                 // BL*DMODEL
  float* buf_xn    = buf_attn  + (size_t)BL * DMODEL;                 // BL*DMODEL
  float* buf_xz    = buf_xn    + (size_t)BL * DMODEL;                 // BL*2*DINNER
  float* buf_uc    = buf_xz    + (size_t)BL * 2 * DINNER;             // BL*DINNER
  float* buf_xdbl  = buf_uc    + (size_t)BL * DINNER;                 // BL*(DTRANK+2*DSTATE)
  float* buf_delta = buf_xdbl  + (size_t)BL * (DTRANK + 2 * DSTATE);  // BL*DINNER
  float* buf_y     = buf_delta + (size_t)BL * DINNER;                 // BL*DINNER

  const long long QKV = 3 * DMODEL;  // 6144

  // 1) h = x @ proj_w^T + proj_b
  gemm(stream, x, DMODEL, proj_w, DMODEL, 1, buf_h, DMODEL,
       proj_b, nullptr, 0, BL, DMODEL, DMODEL, 1.f, /*bias*/1);

  // 2) qkv = h @ qkv_w^T + qkv_b
  gemm(stream, buf_h, DMODEL, qkv_w, DMODEL, 1, buf_qkv, QKV,
       qkv_b, nullptr, 0, BL, QKV, DMODEL, 1.f, 1);

  // 3) scores[b,h] = (Q @ K^T) / sqrt(HDIM)   (batched over 16 (b,h) pairs)
  gemm(stream, buf_qkv, QKV, buf_qkv + DMODEL, QKV, 1, buf_sc, SEQ,
       nullptr, nullptr, 0, SEQ, SEQ, HDIM, 1.f / 16.f, /*scale*/2,
       BATCH * NHEADS, NHEADS,
       (long long)SEQ * QKV, HDIM,          // A offsets (b, h)
       (long long)SEQ * QKV, HDIM,          // W offsets
       (long long)NHEADS * SEQ * SEQ, (long long)SEQ * SEQ);

  // 4) softmax over keys
  softmax_rows<<<BATCH * NHEADS * SEQ, 256, 0, stream>>>(buf_sc, SEQ);

  // 5) ctx[b,h] = P @ V   (W[n,k] = V[k,n] -> ldwn=1, ldwk=QKV, strided template path)
  gemm(stream, buf_sc, SEQ, buf_qkv + 2 * DMODEL, 1, QKV, buf_ctx, DMODEL,
       nullptr, nullptr, 0, SEQ, HDIM, SEQ, 1.f, 0,
       BATCH * NHEADS, NHEADS,
       (long long)NHEADS * SEQ * SEQ, (long long)SEQ * SEQ,
       (long long)SEQ * QKV, HDIM,
       (long long)SEQ * DMODEL, HDIM);

  // 6) attn_out = ctx @ attn_out_w^T + attn_out_b
  gemm(stream, buf_ctx, DMODEL, attn_out_w, DMODEL, 1, buf_attn, DMODEL,
       attn_out_b, nullptr, 0, BL, DMODEL, DMODEL, 1.f, 1);

  // 7) LayerNorm
  layernorm_rows<<<BL, 256, 0, stream>>>(buf_attn, buf_xn, ln_g, ln_b, DMODEL);

  // 8) xz = xn @ in_proj_w^T
  gemm(stream, buf_xn, DMODEL, in_proj_w, DMODEL, 1, buf_xz, 2 * DINNER,
       nullptr, nullptr, 0, BL, 2 * DINNER, DMODEL, 1.f, 0);

  // 9) uc = silu(causal depthwise conv(u) + conv_b)
  conv_silu<<<(BL * DINNER) / 256, 256, 0, stream>>>(buf_xz, conv_w, conv_b, buf_uc);

  // 10) x_dbl = uc @ x_proj_w^T   (N = 160, bounds-checked tiles)
  gemm(stream, buf_uc, DINNER, x_proj_w, DINNER, 1, buf_xdbl, DTRANK + 2 * DSTATE,
       nullptr, nullptr, 0, BL, DTRANK + 2 * DSTATE, DINNER, 1.f, 0);

  // 11) delta = softplus(dt @ dt_proj_w^T + dt_proj_b)   (dt = x_dbl[:, :128])
  gemm(stream, buf_xdbl, DTRANK + 2 * DSTATE, dt_proj_w, DTRANK, 1, buf_delta, DINNER,
       dt_proj_b, nullptr, 0, BL, DINNER, DTRANK, 1.f, /*bias|softplus*/ 1 | 8);

  // 12) selective scan (fused +u*D and *silu(z))
  sel_scan<<<BATCH * (DINNER / 256), 256, 0, stream>>>(
      buf_uc, buf_delta, buf_xdbl, A_log, Dvec, buf_xz, buf_y);

  // 13) out = y @ out_proj_w^T + attn_out
  gemm(stream, buf_y, DINNER, out_proj_w, DINNER, 1, out, DMODEL,
       nullptr, buf_attn, DMODEL, BL, DMODEL, DINNER, 1.f, /*residual*/4);
}
